// fft_attention_30468497998519
// MI455X (gfx1250) — compile-verified, matrix operand order checked by reading, not run on hardware
//
#include <hip/hip_runtime.h>
#include <hip/hip_bf16.h>

// ---------------------------------------------------------------------------
// FFT cross-correlation "attention" for MI455X (gfx1250, wave32, WMMA).
//
//   q = X@Wq.T+bq ; k likewise                  -> bf16 WMMA GEMMs
//       (tiles staged with global_load_async_to_lds_b128 + s_wait_asynccnt)
//   vsum[b,s,h] = values@Wvsum.T + bvsum        -> tiny GEMM (Wv collapsed over e)
//   corr[b,h,e,:] = circ-xcorr(q_series,k_series) via four-step FFT:
//       FFT4096 = F64 x twiddle x F64  (64x64x64 matmuls -> f32 WMMA)
//       (series loaded to LDS via TDM tensor_load_to_lds + s_wait_tensorcnt)
//   out[b,s,e*16+h] = corr[b,h,e,s]*vsum[b,s,h] -> bf16
//   final = out@Wo.T+bo                         -> bf16 WMMA GEMM -> d_out
// ---------------------------------------------------------------------------

typedef __bf16 bf16_t;
typedef __attribute__((ext_vector_type(16))) __bf16 v16bf;
typedef __attribute__((ext_vector_type(8)))  __bf16 v8bf;
typedef __attribute__((ext_vector_type(8)))  float  v8f;
typedef __attribute__((ext_vector_type(2)))  float  v2f;
typedef __attribute__((ext_vector_type(4)))  unsigned int u32x4;
typedef __attribute__((ext_vector_type(8)))  int    i32x8;
typedef __attribute__((ext_vector_type(4)))  int    i32x4;

#define D_MODEL 1024
#define SEQ     4096
#define BATCH   4
#define HEADS   16
#define EDIM    64
#define MROWS   (BATCH * SEQ)          // 16384
#define PI_F    3.14159265358979323846f

// ---------------------------------------------------------------------------
// async global->LDS helpers (CDNA5 ASYNCcnt path, no VGPR staging)
// ---------------------------------------------------------------------------
__device__ inline void async_ld_b128(void* lds_dst, const void* gsrc) {
  unsigned int lo = (unsigned int)(size_t)lds_dst;   // LDS addr = generic[31:0]
  unsigned long long ga = (unsigned long long)(size_t)gsrc;
  asm volatile("global_load_async_to_lds_b128 %0, %1, off"
               :: "v"(lo), "v"(ga) : "memory");
}
__device__ inline void wait_asynccnt0() {
  asm volatile("s_wait_asynccnt 0x0" ::: "memory");
}

// ---------------------------------------------------------------------------
// TDM: 1D f32 tile, global -> LDS (D# per CDNA5 ISA 8.3/8.4; 6-arg builtin)
// ---------------------------------------------------------------------------
__device__ inline void tdm_load_f32_1d(const void* gsrc, void* lds_dst,
                                       unsigned int nelem) {
  unsigned long long ga = (unsigned long long)(size_t)gsrc;
  u32x4 g0;
  g0[0] = 1u;                                         // count=1 user descriptor
  g0[1] = (unsigned int)(size_t)lds_dst;              // lds_addr (bytes)
  g0[2] = (unsigned int)(ga & 0xFFFFFFFFu);           // global_addr[31:0]
  g0[3] = (unsigned int)((ga >> 32) & 0x1FFFFFFu) | (2u << 30);  // [56:32]|type=2
  i32x8 g1;
  g1[0] = (int)(2u << 16);                            // data_size=4B, mask=0
  g1[1] = (int)((nelem & 0xFFFFu) << 16);             // tensor_dim0[15:0]
  g1[2] = (int)(((nelem >> 16) & 0xFFFFu) | (1u << 16)); // td0[31:16]|tensor_dim1=1
  g1[3] = (int)((nelem & 0xFFFFu) << 16);             // tile_dim0
  g1[4] = 0;                                          // tile_dim1/2 unused
  g1[5] = (int)nelem;                                 // tensor_dim0_stride lo
  g1[6] = 0;
  g1[7] = 0;
  i32x4 z4 = {0, 0, 0, 0};
  i32x8 z8 = {0, 0, 0, 0, 0, 0, 0, 0};
  __builtin_amdgcn_tensor_load_to_lds(g0, g1, z4, z4, z8, 0);
}

// ---------------------------------------------------------------------------
// small utility kernels
// ---------------------------------------------------------------------------
__global__ void cvt_f32_to_bf16(const float* __restrict__ src,
                                bf16_t* __restrict__ dst, int n) {
  int i = blockIdx.x * blockDim.x + threadIdx.x;
  if (i < n) dst[i] = (bf16_t)src[i];
}

// Wvsum[h][k] = sum_e Wv[h*64+e][k] ; bvsum[h] = sum_e bv[h*64+e]
__global__ void build_wvsum(const float* __restrict__ Wv,
                            const float* __restrict__ bv,
                            float* __restrict__ Wvsum,
                            float* __restrict__ bvsum) {
  int idx = blockIdx.x * blockDim.x + threadIdx.x;
  if (idx < HEADS * D_MODEL) {
    int h = idx >> 10, k = idx & (D_MODEL - 1);
    float s = 0.f;
#pragma unroll 8
    for (int e = 0; e < EDIM; ++e)
      s += Wv[(size_t)(h * EDIM + e) * D_MODEL + k];
    Wvsum[idx] = s;
  }
  if (idx < HEADS) {
    float s = 0.f;
    for (int e = 0; e < EDIM; ++e) s += bv[idx * EDIM + e];
    bvsum[idx] = s;
  }
}

// vsum[m][h] = values[m][:] . Wvsum[h][:] + bvsum[h]   (N=16, VALU is fine)
__global__ __launch_bounds__(128) void vsum_kernel(
    const float* __restrict__ V, const float* __restrict__ Wvsum,
    const float* __restrict__ bvsum, float* __restrict__ out) {
  __shared__ float rows[8][D_MODEL + 8];
  int m0 = blockIdx.x * 8;
  for (int i = threadIdx.x; i < 8 * D_MODEL; i += 128)
    rows[i >> 10][i & (D_MODEL - 1)] =
        V[(size_t)(m0 + (i >> 10)) * D_MODEL + (i & (D_MODEL - 1))];
  __syncthreads();
  int r = threadIdx.x & 7, h = threadIdx.x >> 3;  // 8 rows x 16 heads
  float acc = bvsum[h];
  const float* wrow = Wvsum + (size_t)h * D_MODEL;
#pragma unroll 4
  for (int k = 0; k < D_MODEL; ++k) acc += rows[r][k] * wrow[k];
  out[(size_t)(m0 + r) * HEADS + h] = acc;
}

// ---------------------------------------------------------------------------
// bf16 WMMA GEMM:  C[M,1024] = A[M,1024] @ W[n][k] + bias
// storeMode 0: C[m*1024+n] (f32)
// storeMode 1: C[((m>>12)*1024+n)*4096 + (m&4095)] (f32, per-series layout)
// 128x128 tile, BK=32, 256 thr = 8 waves, 2x4 WMMA tiles/wave, double-buffered
// via async global->LDS loads.
// ---------------------------------------------------------------------------
__device__ inline v16bf ld_frag16(const bf16_t* p0, const bf16_t* p1) {
  v8bf lo = *(const v8bf*)p0;
  v8bf hi = *(const v8bf*)p1;
  v16bf r;
#pragma unroll
  for (int i = 0; i < 8; ++i) { r[i] = lo[i]; r[8 + i] = hi[i]; }
  return r;
}

__global__ __launch_bounds__(256) void gemm_bf16(
    const bf16_t* __restrict__ A, const bf16_t* __restrict__ Bt,
    const float* __restrict__ bias, float* __restrict__ C, int storeMode) {
  __shared__ bf16_t As[2][128][32];
  __shared__ bf16_t Bs[2][128][32];

  const int tid  = threadIdx.x;
  const int wave = tid >> 5;
  const int lane = tid & 31;
  const int m0 = blockIdx.y * 128;
  const int n0 = blockIdx.x * 128;
  const int wm = wave & 3;   // 32-row group
  const int wn = wave >> 2;  // 64-col group

  v8f acc[2][4];
#pragma unroll
  for (int i = 0; i < 2; ++i)
#pragma unroll
    for (int j = 0; j < 4; ++j) acc[i][j] = (v8f)(0.f);

  auto loadTiles = [&](int kb, int buf) {
#pragma unroll
    for (int i = 0; i < 2; ++i) {
      int idx = tid + i * 256;     // 0..511
      int row = idx >> 2;          // 0..127
      int q4  = idx & 3;           // 16B chunk
      async_ld_b128(&As[buf][row][q4 * 8],
                    A + (size_t)(m0 + row) * D_MODEL + kb * 32 + q4 * 8);
      async_ld_b128(&Bs[buf][row][q4 * 8],
                    Bt + (size_t)(n0 + row) * D_MODEL + kb * 32 + q4 * 8);
      const int kpf = (kb + 2) & (D_MODEL / 32 - 1);  // branchless prefetch
      __builtin_prefetch(A + (size_t)(m0 + row) * D_MODEL + kpf * 32, 0, 0);
      __builtin_prefetch(Bt + (size_t)(n0 + row) * D_MODEL + kpf * 32, 0, 0);
    }
  };

  loadTiles(0, 0);
  wait_asynccnt0();
  __syncthreads();

  const int halfK = (lane >> 4) * 8;  // 0 or 8
  const int lrow  = lane & 15;

  for (int kb = 0; kb < D_MODEL / 32; ++kb) {
    const int buf = kb & 1;
    if (kb + 1 < D_MODEL / 32) loadTiles(kb + 1, buf ^ 1);

    v16bf afrag[2], bfrag[4];
#pragma unroll
    for (int i = 0; i < 2; ++i) {
      int mr = wm * 32 + i * 16 + lrow;
      afrag[i] = ld_frag16(&As[buf][mr][halfK], &As[buf][mr][16 + halfK]);
    }
#pragma unroll
    for (int j = 0; j < 4; ++j) {
      int nr = wn * 64 + j * 16 + lrow;
      bfrag[j] = ld_frag16(&Bs[buf][nr][halfK], &Bs[buf][nr][16 + halfK]);
    }
#pragma unroll
    for (int i = 0; i < 2; ++i)
#pragma unroll
      for (int j = 0; j < 4; ++j)
        acc[i][j] = __builtin_amdgcn_wmma_f32_16x16x32_bf16(
            false, afrag[i], false, bfrag[j], (short)0, acc[i][j], false, false);

    wait_asynccnt0();   // drain this iteration's async fills of buf^1
    __syncthreads();
  }

#pragma unroll
  for (int i = 0; i < 2; ++i) {
#pragma unroll
    for (int j = 0; j < 4; ++j) {
      const int nglob = n0 + wn * 64 + j * 16 + lrow;
      const float bval = bias[nglob];
      const int mbase  = m0 + wm * 32 + i * 16 + (lane >> 4) * 8;
#pragma unroll
      for (int v = 0; v < 8; ++v) {
        const float val = acc[i][j][v] + bval;
        const int m = mbase + v;
        if (storeMode == 0) {
          C[(size_t)m * D_MODEL + nglob] = val;
        } else {
          C[((size_t)(m >> 12) * D_MODEL + nglob) * SEQ + (m & (SEQ - 1))] = val;
        }
      }
    }
  }
}

// ---------------------------------------------------------------------------
// f32 WMMA 64x64x64 complex matmul helper for the four-step FFT.
// A = aRe + i*sa*aIm ; B = bRe + i*sb*bIm ; D = A*B (dIm null -> Re only,
// bIm null -> B real). Planes are 64x64 row-major f32 in LDS.
// 8 waves, each wave owns 2 of the 16 output 16x16 tiles (sharing its A frag).
// ---------------------------------------------------------------------------
__device__ inline v8f wmma4f32(v2f a, v2f b, v8f c) {
  return __builtin_amdgcn_wmma_f32_16x16x4_f32(false, a, false, b, (short)0, c,
                                               false, false);
}

__device__ void cmm64(float* __restrict__ dRe, float* __restrict__ dIm,
                      const float* __restrict__ aRe, const float* __restrict__ aIm,
                      const float* __restrict__ bRe, const float* __restrict__ bIm,
                      float sa, float sb, int tid) {
  const int wave   = tid >> 5;
  const int lane   = tid & 31;
  const int mt     = wave >> 1;          // m-tile 0..3
  const int ntBase = (wave & 1) * 2;     // n-tiles ntBase, ntBase+1
  const int lrow   = lane & 15;
  const int khalf  = (lane >> 4) * 2;    // 0 or 2

  v8f cRe[2], cIm[2];
#pragma unroll
  for (int j = 0; j < 2; ++j) { cRe[j] = (v8f)(0.f); cIm[j] = (v8f)(0.f); }

  const int Mrow = mt * 16 + lrow;
#pragma unroll 4
  for (int k4 = 0; k4 < 16; ++k4) {
    const int K = k4 * 4 + khalf;
    v2f aR = *(const v2f*)(aRe + Mrow * 64 + K);
    v2f aI = (v2f)(0.f);
    if (aIm) { aI = *(const v2f*)(aIm + Mrow * 64 + K); aI *= sa; }
#pragma unroll
    for (int j = 0; j < 2; ++j) {
      const int N = (ntBase + j) * 16 + lrow;
      v2f bR, bI;
      bR.x = bRe[K * 64 + N];
      bR.y = bRe[(K + 1) * 64 + N];
      cRe[j] = wmma4f32(aR, bR, cRe[j]);
      if (bIm) {
        bI.x = bIm[K * 64 + N];
        bI.y = bIm[(K + 1) * 64 + N];
        bI *= sb;
        if (aIm) cRe[j] = wmma4f32(-aI, bI, cRe[j]);   // - sa*sb*aIm*bIm
        if (dIm) cIm[j] = wmma4f32(aR, bI, cIm[j]);    //   sb*aRe*bIm
      }
      if (dIm && aIm) cIm[j] = wmma4f32(aI, bR, cIm[j]);  // sa*aIm*bRe
    }
  }

  const int Mb = mt * 16 + (lane >> 4) * 8;
#pragma unroll
  for (int j = 0; j < 2; ++j) {
    const int N = (ntBase + j) * 16 + lrow;
#pragma unroll
    for (int v = 0; v < 8; ++v) {
      dRe[(Mb + v) * 64 + N] = cRe[j][v];
      if (dIm) dIm[(Mb + v) * 64 + N] = cIm[j][v];
    }
  }
}

// ---------------------------------------------------------------------------
// One block per (b,h,e) series. Four-step FFT(4096)=F64*tw*F64 entirely in LDS.
// 9 planes * 16KB = 144KB dynamic LDS (WGP has 320KB). Series are fetched by
// the Tensor Data Mover (one 16KB 1D descriptor per series).
// ---------------------------------------------------------------------------
__global__ __launch_bounds__(256) void fft_corr_kernel(
    const float* __restrict__ qT, const float* __restrict__ kT,
    const float* __restrict__ vsum, bf16_t* __restrict__ attnout) {
  extern __shared__ float sm[];
  float* Fre = sm + 0 * 4096;
  float* Fim = sm + 1 * 4096;
  float* Tre = sm + 2 * 4096;
  float* Tim = sm + 3 * 4096;
  float* Ure = sm + 4 * 4096;
  float* Uim = sm + 5 * 4096;
  float* Kre = sm + 6 * 4096;
  float* Kim = sm + 7 * 4096;
  float* Xp  = sm + 8 * 4096;

  const int tid = threadIdx.x;
  const int sid = blockIdx.x;        // b*1024 + h*64 + e
  const int b = sid >> 10;
  const int n = sid & 1023;
  const int h = n >> 6;
  const int e = n & 63;

  // TDM: q series -> Xp (wave 0 issues; TENSORcnt drains before the barrier)
  if (tid < 32) tdm_load_f32_1d(qT + (size_t)sid * SEQ, Xp, SEQ);

  // DFT-64 matrix (symmetric), built while the TDM transfer is in flight
  for (int i = tid; i < 4096; i += 256) {
    int r = i >> 6, c = i & 63;
    float s, co;
    __sincosf(-2.0f * PI_F * (float)((r * c) & 63) * (1.0f / 64.0f), &s, &co);
    Fre[i] = co; Fim[i] = s;
  }
  __builtin_amdgcn_s_wait_tensorcnt(0);
  __syncthreads();

  // ---- forward FFT of q: T = F*X ; twiddle ; Q = T*F
  cmm64(Tre, Tim, Fre, Fim, Xp, nullptr, 1.f, 1.f, tid);
  __syncthreads();
  for (int i = tid; i < 4096; i += 256) {
    int c = i >> 6, bb = i & 63;
    float s, co;
    __sincosf(-2.0f * PI_F * (float)(c * bb) * (1.0f / 4096.0f), &s, &co);
    float re = Tre[i], im = Tim[i];
    Tre[i] = re * co - im * s;
    Tim[i] = re * s + im * co;
  }
  __syncthreads();
  cmm64(Ure, Uim, Tre, Tim, Fre, Fim, 1.f, 1.f, tid);

  // TDM: k series -> Xp (safe: Xp dead after q stage1; barrier below orders it)
  __syncthreads();
  if (tid < 32) tdm_load_f32_1d(kT + (size_t)sid * SEQ, Xp, SEQ);
  __builtin_amdgcn_s_wait_tensorcnt(0);
  __syncthreads();

  // ---- forward FFT of k
  cmm64(Tre, Tim, Fre, Fim, Xp, nullptr, 1.f, 1.f, tid);
  __syncthreads();
  for (int i = tid; i < 4096; i += 256) {
    int c = i >> 6, bb = i & 63;
    float s, co;
    __sincosf(-2.0f * PI_F * (float)(c * bb) * (1.0f / 4096.0f), &s, &co);
    float re = Tre[i], im = Tim[i];
    Tre[i] = re * co - im * s;
    Tim[i] = re * s + im * co;
  }
  __syncthreads();
  cmm64(Kre, Kim, Tre, Tim, Fre, Fim, 1.f, 1.f, tid);
  __syncthreads();

  // ---- spectrum product S = Q * conj(K)  -> T planes
  for (int i = tid; i < 4096; i += 256) {
    float qr = Ure[i], qi = Uim[i], kr = Kre[i], ki = Kim[i];
    Tre[i] = qr * kr + qi * ki;
    Tim[i] = qi * kr - qr * ki;
  }
  __syncthreads();

  // ---- inverse FFT: P = S*conj(F) ; conj-twiddle*(1/N) ; x = Re(conj(F)*P)
  cmm64(Ure, Uim, Tre, Tim, Fre, Fim, 1.f, -1.f, tid);
  __syncthreads();
  for (int i = tid; i < 4096; i += 256) {
    int c = i >> 6, bb = i & 63;
    float s, co;
    __sincosf(2.0f * PI_F * (float)(c * bb) * (1.0f / 4096.0f), &s, &co);
    float re = Ure[i], im = Uim[i];
    const float invN = 1.0f / 4096.0f;
    Ure[i] = (re * co - im * s) * invN;
    Uim[i] = (re * s + im * co) * invN;
  }
  __syncthreads();
  cmm64(Xp, nullptr, Fre, Fim, Ure, Uim, -1.f, 1.f, tid);
  __syncthreads();

  // ---- out[b,s,e*16+h] = corr[s] * vsum[b,s,h]  (bf16 for the final GEMM)
  const float* vs = vsum + (size_t)b * SEQ * HEADS + h;
  bf16_t* outp = attnout + (size_t)b * SEQ * D_MODEL + e * HEADS + h;
  for (int s = tid; s < SEQ; s += 256)
    outp[(size_t)s * D_MODEL] = (bf16_t)(Xp[s] * vs[(size_t)s * HEADS]);
}

// ---------------------------------------------------------------------------
extern "C" void kernel_launch(void* const* d_in, const int* in_sizes, int n_in,
                              void* d_out, int out_size, void* d_ws,
                              size_t ws_size, hipStream_t stream) {
  const float* queries = (const float*)d_in[0];
  const float* keys    = (const float*)d_in[1];
  const float* values  = (const float*)d_in[2];
  const float* Wq = (const float*)d_in[3];
  const float* bq = (const float*)d_in[4];
  const float* Wk = (const float*)d_in[5];
  const float* bk = (const float*)d_in[6];
  const float* Wv = (const float*)d_in[7];
  const float* bv = (const float*)d_in[8];
  const float* Wo = (const float*)d_in[9];
  const float* bo = (const float*)d_in[10];
  float* out = (float*)d_out;

  char* wp = (char*)d_ws;
  auto alloc = [&](size_t bytes) {
    char* p = wp;
    wp += (bytes + 255) & ~(size_t)255;
    return p;
  };
  const size_t actE = (size_t)MROWS * D_MODEL;
  bf16_t* qa    = (bf16_t*)alloc(actE * sizeof(bf16_t));
  bf16_t* ka    = (bf16_t*)alloc(actE * sizeof(bf16_t));
  bf16_t* Wqb   = (bf16_t*)alloc((size_t)D_MODEL * D_MODEL * sizeof(bf16_t));
  bf16_t* Wkb   = (bf16_t*)alloc((size_t)D_MODEL * D_MODEL * sizeof(bf16_t));
  bf16_t* Wob   = (bf16_t*)alloc((size_t)D_MODEL * D_MODEL * sizeof(bf16_t));
  float*  Wvs   = (float*)alloc((size_t)HEADS * D_MODEL * sizeof(float));
  float*  bvs   = (float*)alloc(HEADS * sizeof(float));
  float*  qTbuf = (float*)alloc((size_t)BATCH * D_MODEL * SEQ * sizeof(float));
  float*  kTbuf = (float*)alloc((size_t)BATCH * D_MODEL * SEQ * sizeof(float));
  float*  vsumb = (float*)alloc((size_t)MROWS * HEADS * sizeof(float));
  bf16_t* attno = (bf16_t*)alloc(actE * sizeof(bf16_t));

  const int nAct = MROWS * D_MODEL;
  const int nW   = D_MODEL * D_MODEL;
  cvt_f32_to_bf16<<<(nAct + 255) / 256, 256, 0, stream>>>(queries, qa, nAct);
  cvt_f32_to_bf16<<<(nAct + 255) / 256, 256, 0, stream>>>(keys, ka, nAct);
  cvt_f32_to_bf16<<<(nW + 255) / 256, 256, 0, stream>>>(Wq, Wqb, nW);
  cvt_f32_to_bf16<<<(nW + 255) / 256, 256, 0, stream>>>(Wk, Wkb, nW);
  cvt_f32_to_bf16<<<(nW + 255) / 256, 256, 0, stream>>>(Wo, Wob, nW);
  build_wvsum<<<(HEADS * D_MODEL + 255) / 256, 256, 0, stream>>>(Wv, bv, Wvs, bvs);
  vsum_kernel<<<MROWS / 8, 128, 0, stream>>>(values, Wvs, bvs, vsumb);

  dim3 gg(D_MODEL / 128, MROWS / 128);
  gemm_bf16<<<gg, 256, 0, stream>>>(qa, Wqb, bq, qTbuf, 1);
  gemm_bf16<<<gg, 256, 0, stream>>>(ka, Wkb, bk, kTbuf, 1);

  fft_corr_kernel<<<BATCH * D_MODEL, 256, 9 * 4096 * sizeof(float), stream>>>(
      qTbuf, kTbuf, vsumb, attno);

  gemm_bf16<<<gg, 256, 0, stream>>>(attno, Wob, bo, out, 0);
}
